// EnhancedMemoryEfficientMoE_37864431681943
// MI455X (gfx1250) — matrix operand order repaired and checked by reading, not compiled, not run on hardware
//
#include <hip/hip_runtime.h>
#include <math.h>

// ---------------- static problem config ----------------
constexpr int Bb   = 2;
constexpr int Ss   = 4096;
constexpr int Hh_  = 2048;
constexpr int Ee   = 16;
constexpr int BSb  = 512;
constexpr int NBb  = 8;
constexpr int Tt   = Bb * Ss;        // 8192 tokens
constexpr int DTOK = 1024;
constexpr int DBLK = 256;
constexpr int FBl  = Bb * NBb;       // 16 flat blocks
constexpr int MAXTOK = 1228;         // int(4096*0.3)
constexpr int NWG2 = 32;             // k_apply workgroups

// ---------------- workspace layout (bytes) ----------------
constexpr size_t OFF_HBF   = 0;                                   // hidden bf16  [Tt][H]
constexpr size_t SZ_HBF    = (size_t)Tt * Hh_ * 2;                // 33.5 MB
constexpr size_t OFF_W1T   = OFF_HBF + SZ_HBF;                    // tr_w1^T bf16 [DTOK][H]
constexpr size_t SZ_W1T    = (size_t)DTOK * Hh_ * 2;              // 4 MB
constexpr size_t OFF_W2T   = OFF_W1T + SZ_W1T;                    // tr_w2^T bf16 [E][DTOK]
constexpr size_t SZ_W2T    = (size_t)Ee * DTOK * 2;
constexpr size_t OFF_HMID  = OFF_W2T + SZ_W2T;                    // gelu(h) bf16 [Tt][DTOK]
constexpr size_t SZ_HMID   = (size_t)Tt * DTOK * 2;               // 16.8 MB
constexpr size_t OFF_BREP  = OFF_HMID + SZ_HMID;                  // block_repr f32 [16][H]
constexpr size_t SZ_BREP   = (size_t)FBl * Hh_ * 4;
constexpr size_t OFF_XWS   = OFF_BREP + SZ_BREP;                  // block router x f32 [16][256]
constexpr size_t SZ_XWS    = (size_t)FBl * DBLK * 4;
constexpr size_t OFF_MODES = OFF_XWS + SZ_XWS;                    // int[16]
constexpr size_t OFF_PAY   = OFF_MODES + 256;                     // f32[16][16]
constexpr size_t OFF_PART  = OFF_PAY + 2048;                      // f32[32][16] usage partials

// ---------------- types for WMMA ----------------
typedef float          v8f   __attribute__((ext_vector_type(8)));
typedef __bf16         v16bf __attribute__((ext_vector_type(16)));
typedef unsigned short v8u   __attribute__((ext_vector_type(8)));
typedef unsigned short v16u  __attribute__((ext_vector_type(16)));

union BF16x16 { v16u u; v16bf b; };

__device__ __forceinline__ unsigned short f2bf(float f) {
    unsigned u = __float_as_uint(f);
    unsigned r = (u + 0x7FFFu + ((u >> 16) & 1u)) >> 16;   // round-to-nearest-even
    return (unsigned short)r;
}

__device__ __forceinline__ float gelu_exact(float x) {
    return 0.5f * x * (1.0f + erff(x * 0.70710678118654752f));
}

// A fragment (16x32 bf16) from row-major [.., K]:
//  lane group hi = lane>>4 holds K = 16hi + i (i=0..15 contiguous), row = base + (lane&15)
__device__ __forceinline__ v16bf load_frag_a(const unsigned short* base, int row, int K,
                                             int k0, int lane) {
    int hi = lane >> 4;
    const unsigned short* p = base + (size_t)row * K + k0 + 16 * hi;
    v8u lo = *(const v8u*)p;
    v8u hv = *(const v8u*)(p + 8);
    BF16x16 f;
#pragma unroll
    for (int i = 0; i < 8; ++i) { f.u[i] = lo[i]; f.u[i + 8] = hv[i]; }
    return f.b;
}

// B fragment (32x16 bf16) from transposed [N][K]:
//  lane group hi holds K = 16hi + i (i=0..15 contiguous), col N = nbase + (lane&15)
__device__ __forceinline__ v16bf load_frag_b(const unsigned short* baseT, int n, int K,
                                             int k0, int lane) {
    int hi = lane >> 4;
    const unsigned short* p = baseT + (size_t)n * K + k0 + 16 * hi;
    v8u lo = *(const v8u*)p;
    v8u hv = *(const v8u*)(p + 8);
    BF16x16 f;
#pragma unroll
    for (int i = 0; i < 8; ++i) { f.u[i] = lo[i]; f.u[i + 8] = hv[i]; }
    return f.b;
}

// ---------------- conversion / transpose kernels ----------------
__global__ __launch_bounds__(256) void k_cvt(const float* __restrict__ in,
                                             unsigned short* __restrict__ out, int n) {
    for (int i = blockIdx.x * blockDim.x + threadIdx.x; i < n; i += gridDim.x * blockDim.x)
        out[i] = f2bf(in[i]);
}

__global__ __launch_bounds__(256) void k_tw1(const float* __restrict__ w1,
                                             unsigned short* __restrict__ w1t) {
    int i = blockIdx.x * blockDim.x + threadIdx.x;          // over DTOK*H
    if (i < DTOK * Hh_) {
        int n = i >> 11, k = i & 2047;
        w1t[i] = f2bf(w1[(size_t)k * DTOK + n]);
    }
}

__global__ __launch_bounds__(256) void k_tw2(const float* __restrict__ w2,
                                             unsigned short* __restrict__ w2t) {
    int i = blockIdx.x * blockDim.x + threadIdx.x;          // over E*DTOK
    if (i < Ee * DTOK) {
        int n = i >> 10, k = i & 1023;
        w2t[i] = f2bf(w2[(size_t)k * Ee + n]);
    }
}

// ---------------- block_repr: mean over BS tokens ----------------
__global__ __launch_bounds__(256) void k_blockrepr(const float* __restrict__ hs,
                                                   float* __restrict__ brep) {
    int blk = blockIdx.x;                // 0..15 flat (b,nb); row0 = blk*512
    int t = threadIdx.x;
    int row0 = blk * BSb;
#pragma unroll 1
    for (int c = 0; c < 8; ++c) {
        int col = t + 256 * c;
        float s = 0.f;
        for (int i = 0; i < BSb; ++i)
            s += hs[(size_t)(row0 + i) * Hh_ + col];
        brep[(size_t)blk * Hh_ + col] = s * (1.0f / (float)BSb);
    }
}

// ---------------- block router GEMM1 (fp32, exact) ----------------
__global__ __launch_bounds__(256) void k_bgemm(const float* __restrict__ brep,
                                               const float* __restrict__ w1,
                                               const float* __restrict__ b1,
                                               float* __restrict__ xws) {
    int blk = blockIdx.x, d = threadIdx.x;
    const float* br = brep + (size_t)blk * Hh_;
    float s = b1[d];
    for (int k = 0; k < Hh_; ++k)
        s += br[k] * w1[(size_t)k * DBLK + d];
    xws[blk * DBLK + d] = s;
}

// ---------------- block router: LN/GELU/GEMM2/softmax/entropy + budget logic ----------------
__global__ __launch_bounds__(256) void k_router(const float* __restrict__ xin,
                                                const float* __restrict__ g,
                                                const float* __restrict__ beta,
                                                const float* __restrict__ w2,
                                                const float* __restrict__ b2,
                                                int* __restrict__ modes,
                                                float* __restrict__ payload) {
    __shared__ float xs[FBl][DBLK];
    __shared__ float lg[FBl][Ee];
    __shared__ float ent[FBl];
    __shared__ float pay[FBl][Ee];
    int tid = threadIdx.x;

    for (int i = tid; i < FBl * DBLK; i += 256) xs[i >> 8][i & 255] = xin[i];
    __syncthreads();

    if (tid < FBl) {  // LayerNorm + exact GELU per block row
        float mu = 0.f;
        for (int d = 0; d < DBLK; ++d) mu += xs[tid][d];
        mu *= (1.0f / DBLK);
        float var = 0.f;
        for (int d = 0; d < DBLK; ++d) { float dd = xs[tid][d] - mu; var += dd * dd; }
        var *= (1.0f / DBLK);
        float is = rsqrtf(var + 1e-5f);
        for (int d = 0; d < DBLK; ++d) {
            float y = (xs[tid][d] - mu) * is * g[d] + beta[d];
            xs[tid][d] = gelu_exact(y);
        }
    }
    __syncthreads();

    {   // GEMM2: 16x256 @ 256x16 — one (m,e) per thread
        int m = tid >> 4, e = tid & 15;
        float s = b2[e];
        for (int d = 0; d < DBLK; ++d) s += xs[m][d] * w2[d * Ee + e];
        lg[m][e] = s;
    }
    __syncthreads();

    if (tid < FBl) {  // softmax, entropy, argmax payload
        float mx = -1e30f;
        for (int e = 0; e < Ee; ++e) mx = fmaxf(mx, lg[tid][e]);
        float p[Ee], s = 0.f;
        for (int e = 0; e < Ee; ++e) { p[e] = expf(lg[tid][e] - mx); s += p[e]; }
        float inv = 1.0f / s, en = 0.f, pm = -1.f;
        int am = 0;
        for (int e = 0; e < Ee; ++e) {
            float pp = p[e] * inv;
            float q = pp + 1e-10f;
            en -= q * logf(q);
            if (pp > pm) { pm = pp; am = e; }
            p[e] = pp;
        }
        ent[tid] = en / logf((float)Ee);
        for (int e = 0; e < Ee; ++e) pay[tid][e] = (e == am) ? pm : 0.f;
    }
    __syncthreads();

    if (tid == 0) {   // serial budget logic (16 blocks)
        bool oh[FBl]; int nh = 0;
        for (int f = 0; f < FBl; ++f) { oh[f] = ent[f] > 0.6f; nh += oh[f] ? 1 : 0; }
        bool need = (nh * BSb > MAXTOK) && (nh > 0) && (2 < nh);
        float m1 = -INFINITY, m2 = -INFINITY;
        for (int f = 0; f < FBl; ++f) {
            if (oh[f]) {
                float v = ent[f];
                if (v > m1) { m2 = m1; m1 = v; }
                else if (v > m2) { m2 = v; }
                else if (v == m1) { m2 = v; }   // duplicate of max -> 2nd sorted value
            }
        }
        int inc = 0;
        for (int f = 0; f < FBl; ++f) {
            bool adj = need ? (ent[f] > m2) : oh[f];
            inc += adj ? BSb : 0;
            bool fits = adj && (inc <= MAXTOK);
            bool bm = (!oh[f]) || (adj && !fits);
            modes[f] = fits ? 2 : (bm ? 1 : 0);
        }
        for (int i = 0; i < FBl * Ee; ++i) payload[i] = pay[i >> 4][i & 15];
    }
}

// ---------------- token router GEMM1 + GELU (bf16 WMMA) ----------------
// grid (32, 16) x 256 threads. Wave tile 32x64 (2 m-tiles x 4 n-tiles):
// 12 b128 loads per 8 WMMAs. Workgroup covers M=256, N=64.
__global__ __launch_bounds__(256) void k_gemm1(const unsigned short* __restrict__ A,
                                               const unsigned short* __restrict__ Bt,
                                               const float* __restrict__ b1,
                                               unsigned short* __restrict__ Hm) {
    int lane = threadIdx.x & 31, wave = threadIdx.x >> 5;
    int ln = lane & 15, hi = lane >> 4;
    int m0 = (blockIdx.x * 8 + wave) * 2;    // first of two m-tiles (0..510)
    int n0 = blockIdx.y * 64;
    int rowA0 = m0 * 16 + ln;
    int rowA1 = rowA0 + 16;

    v8f acc[2][4];
#pragma unroll
    for (int t = 0; t < 2; ++t)
#pragma unroll
        for (int j = 0; j < 4; ++j)
#pragma unroll
            for (int v = 0; v < 8; ++v) acc[t][j][v] = 0.f;

#pragma unroll 2
    for (int k0 = 0; k0 < Hh_; k0 += 32) {
        v16bf a0 = load_frag_a(A, rowA0, Hh_, k0, lane);
        v16bf a1 = load_frag_a(A, rowA1, Hh_, k0, lane);
#pragma unroll
        for (int j = 0; j < 4; ++j) {
            v16bf bf = load_frag_b(Bt, n0 + 16 * j + ln, Hh_, k0, lane);
            acc[0][j] = __builtin_amdgcn_wmma_f32_16x16x32_bf16(
                false, a0, false, bf, (short)0, acc[0][j], false, false);
            acc[1][j] = __builtin_amdgcn_wmma_f32_16x16x32_bf16(
                false, a1, false, bf, (short)0, acc[1][j], false, false);
        }
    }

    float bn[4];
#pragma unroll
    for (int j = 0; j < 4; ++j) bn[j] = b1[n0 + 16 * j + ln];

#pragma unroll
    for (int t = 0; t < 2; ++t) {
#pragma unroll
        for (int j = 0; j < 4; ++j) {
            int col = n0 + 16 * j + ln;
#pragma unroll
            for (int v = 0; v < 8; ++v) {
                int row = (m0 + t) * 16 + v + 8 * hi;
                float x = gelu_exact(acc[t][j][v] + bn[j]);
                Hm[(size_t)row * DTOK + col] = f2bf(x);
            }
        }
    }
}

// ---------------- token GEMM2 + softmax + mode apply + usage partials ----------------
// grid 32 x 256 threads; 2 m-tiles per wave (16 tiles / 256 rows per WG).
__global__ __launch_bounds__(256) void k_apply(const unsigned short* __restrict__ Hm,
                                               const unsigned short* __restrict__ W2T,
                                               const float* __restrict__ b2,
                                               const int* __restrict__ modes,
                                               const float* __restrict__ payload,
                                               float* __restrict__ out,
                                               float* __restrict__ partial) {
    __shared__ float lt[16][16][16];
    __shared__ float red[256][16];
    int tid = threadIdx.x;
    int lane = tid & 31, wave = tid >> 5;
    int ln = lane & 15, hi = lane >> 4;
    int mt0 = (blockIdx.x * 8 + wave) * 2;   // first of two m-tiles (0..510)
    int rowA0 = mt0 * 16 + ln;
    int rowA1 = rowA0 + 16;

    v8f acc[2];
#pragma unroll
    for (int t = 0; t < 2; ++t)
#pragma unroll
        for (int v = 0; v < 8; ++v) acc[t][v] = 0.f;

#pragma unroll 2
    for (int k0 = 0; k0 < DTOK; k0 += 32) {
        v16bf a0 = load_frag_a(Hm, rowA0, DTOK, k0, lane);
        v16bf a1 = load_frag_a(Hm, rowA1, DTOK, k0, lane);
        v16bf bf = load_frag_b(W2T, ln, DTOK, k0, lane);
        acc[0] = __builtin_amdgcn_wmma_f32_16x16x32_bf16(
            false, a0, false, bf, (short)0, acc[0], false, false);
        acc[1] = __builtin_amdgcn_wmma_f32_16x16x32_bf16(
            false, a1, false, bf, (short)0, acc[1], false, false);
    }

#pragma unroll
    for (int t = 0; t < 2; ++t)
#pragma unroll
        for (int v = 0; v < 8; ++v) lt[wave * 2 + t][v + 8 * hi][ln] = acc[t][v];
    __syncthreads();

    {   // one token row per thread (256 rows per WG)
        int ltile = tid >> 4, r = tid & 15;
        int row = (blockIdx.x * 16 + ltile) * 16 + r;
        float lgv[Ee];
        float mx = -1e30f;
#pragma unroll
        for (int e = 0; e < Ee; ++e) { lgv[e] = lt[ltile][r][e] + b2[e]; mx = fmaxf(mx, lgv[e]); }
        float s = 0.f;
#pragma unroll
        for (int e = 0; e < Ee; ++e) { lgv[e] = expf(lgv[e] - mx); s += lgv[e]; }
        float inv = 1.0f / s;
        int bb = row >> 12;                 // batch
        int sp = row & (Ss - 1);
        int fb = bb * NBb + (sp >> 9);
        int mode = modes[fb];
#pragma unroll
        for (int e = 0; e < Ee; ++e) {
            float wgt = (mode == 2) ? lgv[e] * inv
                       : (mode == 1) ? payload[fb * Ee + e]
                                     : 0.f;
            out[(size_t)row * Ee + e] = wgt;
            red[tid][e] = wgt;
        }
    }
    __syncthreads();

    // deterministic tree reduction of 256 per-row usage vectors
    for (int st = 128; st >= 1; st >>= 1) {
        if (tid < st) {
#pragma unroll
            for (int e = 0; e < Ee; ++e) red[tid][e] += red[tid + st][e];
        }
        __syncthreads();
    }
    if (tid < Ee) partial[blockIdx.x * Ee + tid] = red[0][tid];
}

// ---------------- load-balance loss ----------------
__global__ void k_loss(const float* __restrict__ partial, float* __restrict__ out) {
    if (threadIdx.x == 0 && blockIdx.x == 0) {
        float loss = 0.f;
        const float tgt = 1.0f / (float)Ee;
        for (int e = 0; e < Ee; ++e) {
            float u = 0.f;
            for (int wg = 0; wg < NWG2; ++wg) u += partial[wg * Ee + e];
            u *= (1.0f / (float)Tt);
            loss += tgt * logf(tgt / (u + 1e-10f));
        }
        out[(size_t)Tt * Ee] = loss;
    }
}

// ---------------- launch ----------------
extern "C" void kernel_launch(void* const* d_in, const int* in_sizes, int n_in,
                              void* d_out, int out_size, void* d_ws, size_t ws_size,
                              hipStream_t stream) {
    const float* hs      = (const float*)d_in[0];
    const float* br_w1   = (const float*)d_in[1];
    const float* br_b1   = (const float*)d_in[2];
    const float* br_g    = (const float*)d_in[3];
    const float* br_beta = (const float*)d_in[4];
    const float* br_w2   = (const float*)d_in[5];
    const float* br_b2   = (const float*)d_in[6];
    const float* tr_w1   = (const float*)d_in[7];
    const float* tr_b1   = (const float*)d_in[8];
    const float* tr_w2   = (const float*)d_in[9];
    const float* tr_b2   = (const float*)d_in[10];

    char* ws = (char*)d_ws;
    unsigned short* hBF  = (unsigned short*)(ws + OFF_HBF);
    unsigned short* W1T  = (unsigned short*)(ws + OFF_W1T);
    unsigned short* W2T  = (unsigned short*)(ws + OFF_W2T);
    unsigned short* Hmid = (unsigned short*)(ws + OFF_HMID);
    float* brep    = (float*)(ws + OFF_BREP);
    float* xws     = (float*)(ws + OFF_XWS);
    int*   modes   = (int*)(ws + OFF_MODES);
    float* payload = (float*)(ws + OFF_PAY);
    float* partial = (float*)(ws + OFF_PART);
    float* out     = (float*)d_out;

    k_cvt<<<2048, 256, 0, stream>>>(hs, hBF, Tt * Hh_);
    k_tw1<<<(DTOK * Hh_ + 255) / 256, 256, 0, stream>>>(tr_w1, W1T);
    k_tw2<<<(Ee * DTOK + 255) / 256, 256, 0, stream>>>(tr_w2, W2T);
    k_blockrepr<<<FBl, 256, 0, stream>>>(hs, brep);
    k_bgemm<<<FBl, DBLK, 0, stream>>>(brep, br_w1, br_b1, xws);
    k_router<<<1, 256, 0, stream>>>(xws, br_g, br_beta, br_w2, br_b2, modes, payload);
    k_gemm1<<<dim3(32, 16), 256, 0, stream>>>(hBF, W1T, tr_b1, Hmid);
    k_apply<<<NWG2, 256, 0, stream>>>(Hmid, W2T, tr_b2, modes, payload, out, partial);
    k_loss<<<1, 32, 0, stream>>>(partial, out);
}
